// MultiHeadAttention_1039382086029
// MI455X (gfx1250) — matrix operand order repaired and checked by reading, not compiled, not run on hardware
//
#include <hip/hip_runtime.h>
#include <hip/hip_bf16.h>

typedef __attribute__((ext_vector_type(16))) __bf16 v16bf;
typedef __attribute__((ext_vector_type(8)))  float  v8f;

#define BSZ 32
#define IH 28
#define IW 28
#define QL 32
#define DIM 512
#define NH 8
#define DKH 64
#define NG (IH*IW)          // 784 grid tokens
#define SEQ (NG+QL)         // 816 total tokens

static_assert(SEQ == 816 && NG == 784, "shape check");

// ---------------- WMMA fragment helpers (gfx1250, wave32) ----------------

union FragBF { v16bf v; uint4 u[2]; };

__device__ inline v8f zero8() {
  v8f z;
#pragma unroll
  for (int i = 0; i < 8; ++i) z[i] = 0.f;
  return z;
}

// A-matrix 16x32 bf16 fragment from row-major tile (stride ld in elems).
// Lane l<16 holds row M=l, K={0..7,16..23}; lane l>=16 holds row M=l-16, K={8..15,24..31}.
__device__ inline v16bf frag_a(const __bf16* base, int ld) {
  unsigned lane = threadIdx.x & 31u;
  unsigned r = lane & 15u, hs = lane >> 4u;
  const __bf16* p = base + (size_t)r * ld + 8u * hs;
  FragBF f;
  f.u[0] = *reinterpret_cast<const uint4*>(p);
  f.u[1] = *reinterpret_cast<const uint4*>(p + 16);
  return f.v;
}

// B-matrix 32x16 bf16 fragment from row-major *transposed* tile Bt[n][k]
// (stride ld). Lanes 0-15 hold K={0..7,16..23} of col n; lanes 16-31 hold K={8..15,24..31}.
__device__ inline v16bf frag_b(const __bf16* base, int ld) {
  unsigned lane = threadIdx.x & 31u;
  unsigned n = lane & 15u, hs = lane >> 4u;
  const __bf16* p = base + (size_t)n * ld + 8u * hs;
  FragBF f;
  f.u[0] = *reinterpret_cast<const uint4*>(p);
  f.u[1] = *reinterpret_cast<const uint4*>(p + 16);
  return f.v;
}

#define WMMA_BF16(a, b, c) \
  __builtin_amdgcn_wmma_f32_16x16x32_bf16(false, (a), false, (b), (short)0, (c), false, false)

// ---------------- Async global -> LDS staging (CDNA5) ----------------
// GLOBAL_LOAD_ASYNC_TO_LDS_B128: 16 bytes/lane, LDS write with no VGPR data
// path, tracked by ASYNCcnt.  Low 32 bits of a flat shared-aperture address
// are the LDS byte offset (ISA aperture table), used as the VDST operand.

__device__ __forceinline__ void async_b128(void* lds_p, const void* gp) {
  unsigned l = (unsigned)(unsigned long long)lds_p;
  asm volatile("global_load_async_to_lds_b128 %0, %1, off"
               :: "v"(l), "v"(gp) : "memory");
}
__device__ __forceinline__ void wait_async0() {
  asm volatile("s_wait_asynccnt 0x0" ::: "memory");
}

// xor-lane data exchange via ds_swizzle (group-of-32 mode), wave32-safe.
template <int XM>
__device__ inline float swz_xor(float v) {
  int x = __builtin_amdgcn_ds_swizzle(__float_as_int(v), (XM << 10) | 0x1f);
  return __int_as_float(x);
}
__device__ inline float redmax16(float v) {
  v = fmaxf(v, swz_xor<1>(v)); v = fmaxf(v, swz_xor<2>(v));
  v = fmaxf(v, swz_xor<4>(v)); v = fmaxf(v, swz_xor<8>(v));
  return v;
}
__device__ inline float redsum16(float v) {
  v += swz_xor<1>(v); v += swz_xor<2>(v);
  v += swz_xor<4>(v); v += swz_xor<8>(v);
  return v;
}

// ---------------- Kernel 1: cast input fp32 -> bf16 ----------------

__global__ __launch_bounds__(256) void cast_input(const float* __restrict__ x,
                                                  __bf16* __restrict__ y) {
  size_t n = (size_t)BSZ * SEQ * DIM;
  for (size_t i = blockIdx.x * (size_t)blockDim.x + threadIdx.x; i < n;
       i += (size_t)gridDim.x * blockDim.x)
    y[i] = (__bf16)x[i];
}

// ---------------- Kernel 2: pack weights to bf16 ----------------
// Wc[proj][tap][o][i] = conv_w[o][i][tap] ; Wl[proj][o][i] ; Wo[o][i]

__global__ __launch_bounds__(256) void pack_weights(
    const float* wk, const float* wq, const float* wv,
    const float* lk, const float* lq, const float* lv,
    const float* wo,
    __bf16* __restrict__ Wc, __bf16* __restrict__ Wl, __bf16* __restrict__ Wo) {
  const float* cw[3] = {wk, wq, wv};
  const float* lw[3] = {lk, lq, lv};
  size_t stride = (size_t)gridDim.x * blockDim.x;
  size_t t0 = blockIdx.x * (size_t)blockDim.x + threadIdx.x;
  size_t nc = (size_t)3 * 9 * DIM * DIM;
  for (size_t i = t0; i < nc; i += stride) {
    size_t r = i;
    int ic = (int)(r % DIM); r /= DIM;
    int o  = (int)(r % DIM); r /= DIM;
    int tp = (int)(r % 9);   r /= 9;
    int pj = (int)r;
    Wc[i] = (__bf16)cw[pj][(size_t)o * DIM * 9 + (size_t)ic * 9 + tp];
  }
  size_t nl = (size_t)3 * DIM * DIM;
  for (size_t i = t0; i < nl; i += stride) {
    int pj = (int)(i / ((size_t)DIM * DIM));
    size_t r = i % ((size_t)DIM * DIM);
    Wl[i] = (__bf16)lw[pj][r];
  }
  size_t no = (size_t)DIM * DIM;
  for (size_t i = t0; i < no; i += stride) Wo[i] = (__bf16)wo[i];
}

// ---------------- Kernel 3: question-token linear projections ----------------
// One wave per 16x16 output tile. M = 32 batches * 32 tokens, N = 512, K = 512.

__global__ __launch_bounds__(32) void qlin(
    const __bf16* __restrict__ Xb, const __bf16* __restrict__ Wl,
    const float* __restrict__ bk, const float* __restrict__ bq,
    const float* __restrict__ bv,
    __bf16* __restrict__ Kb, __bf16* __restrict__ Qb, __bf16* __restrict__ Vt) {
  int mt = blockIdx.x;   // 0..63
  int nt = blockIdx.y;   // 0..31
  int pj = blockIdx.z;   // 0:K 1:Q 2:V
  int b = mt >> 1;
  int s0 = NG + ((mt & 1) << 4);
  const __bf16* Ab = Xb + ((size_t)b * SEQ + s0) * DIM;
  const __bf16* Bb = Wl + ((size_t)pj * DIM + nt * 16) * DIM;
  v8f c = zero8();
#pragma unroll 4
  for (int kk = 0; kk < DIM; kk += 32) {
    v16bf a  = frag_a(Ab + kk, DIM);
    v16bf bm = frag_b(Bb + kk, DIM);   // W[o][i] row-major == Bt[n][k]
    c = WMMA_BF16(a, bm, c);
  }
  const float* bias = (pj == 0) ? bk : ((pj == 1) ? bq : bv);
  unsigned lane = threadIdx.x & 31u, n = lane & 15u, hs = lane >> 4u;
#pragma unroll
  for (int r = 0; r < 8; ++r) {
    int srow = s0 + r + 8 * (int)hs;
    int o = nt * 16 + (int)n;
    float v = c[r] + bias[o];
    if (pj == 0)      Kb[((size_t)b * SEQ + srow) * DIM + o] = (__bf16)v;
    else if (pj == 1) Qb[((size_t)b * SEQ + srow) * DIM + o] = (__bf16)v;
    else Vt[(((size_t)b * NH + (o >> 6)) * DKH + (o & 63)) * SEQ + srow] = (__bf16)v;
  }
}

// ---------------- Kernel 4: conv3x3 QKV as implicit GEMM ----------------
// Block: 224 threads (7 waves). Tile: 112 positions x 64 channels.
// Loop taps (9) x K-chunks (16 of 32): async-stage shifted A + per-tap B in
// LDS (zero-fill borders with plain DS stores), then WMMA.

__global__ __launch_bounds__(224) void convqkv(
    const __bf16* __restrict__ Xb, const __bf16* __restrict__ Wc,
    __bf16* __restrict__ Kb, __bf16* __restrict__ Qb, __bf16* __restrict__ Vt) {
  __shared__ __attribute__((aligned(16))) __bf16 sA[112 * 32];
  __shared__ __attribute__((aligned(16))) __bf16 sB[64 * 32];
  int mt = blockIdx.x;           // 0..6
  int nt = blockIdx.y;           // 0..7
  int bz = blockIdx.z;           // 0..95
  int pj = bz / BSZ;
  int b  = bz % BSZ;
  int wave = (int)(threadIdx.x >> 5);
  const __bf16* Xg = Xb + (size_t)b * SEQ * DIM;  // grid tokens 0..783
  v8f acc[4] = {zero8(), zero8(), zero8(), zero8()};

  for (int tap = 0; tap < 9; ++tap) {
    int dy = tap / 3 - 1, dx = tap % 3 - 1;
    int off = dy * IW + dx;
    const __bf16* Bsrc = Wc + (((size_t)pj * 9 + tap) * DIM + nt * 64) * DIM;
    for (int kc = 0; kc < 16; ++kc) {
      int kk = kc * 32;
      __syncthreads();
      // stage A: 112 rows x 32 cols bf16, border rows zero-filled
      for (int idx = (int)threadIdx.x; idx < 448; idx += 224) {
        int row = idx >> 2, ch = idx & 3;
        int pos = mt * 112 + row;
        int y = pos / IW, x = pos % IW;
        int yy = y + dy, xx = x + dx;
        __bf16* dst = sA + row * 32 + ch * 8;
        if (yy >= 0 && yy < IH && xx >= 0 && xx < IW) {
          async_b128(dst, Xg + (size_t)(pos + off) * DIM + kk + ch * 8);
        } else {
          uint4 z = {0u, 0u, 0u, 0u};
          *reinterpret_cast<uint4*>(dst) = z;
        }
      }
      // stage B: 64 rows x 32 cols
      for (int idx = (int)threadIdx.x; idx < 256; idx += 224) {
        int row = idx >> 2, ch = idx & 3;
        async_b128(sB + row * 32 + ch * 8, Bsrc + (size_t)row * DIM + kk + ch * 8);
      }
      wait_async0();
      __syncthreads();
      v16bf a = frag_a(sA + wave * 16 * 32, 32);
#pragma unroll
      for (int ns = 0; ns < 4; ++ns) {
        v16bf bm = frag_b(sB + ns * 16 * 32, 32);
        acc[ns] = WMMA_BF16(a, bm, acc[ns]);
      }
    }
  }
  unsigned lane = threadIdx.x & 31u, n = lane & 15u, hs = lane >> 4u;
#pragma unroll
  for (int ns = 0; ns < 4; ++ns)
#pragma unroll
    for (int r = 0; r < 8; ++r) {
      int pos = mt * 112 + wave * 16 + r + 8 * (int)hs;
      int o = nt * 64 + ns * 16 + (int)n;
      float v = acc[ns][r];
      if (pj == 0)      Kb[((size_t)b * SEQ + pos) * DIM + o] = (__bf16)v;
      else if (pj == 1) Qb[((size_t)b * SEQ + pos) * DIM + o] = (__bf16)v;
      else Vt[(((size_t)b * NH + (o >> 6)) * DKH + (o & 63)) * SEQ + pos] = (__bf16)v;
    }
}

// ---------------- Kernel 5: flash attention per (b, h) ----------------
// Block: 96 threads (3 waves), each wave owns 16 query rows; 17*3 = 51 q-tiles.
// Stream 32-key blocks: async-stage K/V tiles, scores via WMMA, online softmax
// (ds_swizzle row reductions), P -> LDS -> A-fragment, O += P @ V via WMMA.

__global__ __launch_bounds__(96) void attn(
    const __bf16* __restrict__ Qb, const __bf16* __restrict__ Kb,
    const __bf16* __restrict__ Vt, __bf16* __restrict__ Ob) {
  __shared__ __attribute__((aligned(16))) __bf16 sK[32 * 64];      // [key][d]
  __shared__ __attribute__((aligned(16))) __bf16 sV[64 * 32];      // [d][key]
  __shared__ __attribute__((aligned(16))) __bf16 sP[3][16 * 32];   // per-wave P
  int b = blockIdx.y / NH, h = blockIdx.y % NH;
  int wave = (int)(threadIdx.x >> 5);
  int qt = blockIdx.x * 3 + wave;                // 0..50
  unsigned lane = threadIdx.x & 31u, n = lane & 15u, hs = lane >> 4u;

  const __bf16* qbase = Qb + ((size_t)b * SEQ + qt * 16) * DIM + h * DKH;
  v16bf aq0 = frag_a(qbase, DIM);
  v16bf aq1 = frag_a(qbase + 32, DIM);

  const __bf16* kbase = Kb + (size_t)b * SEQ * DIM + h * DKH;
  const __bf16* vbase = Vt + ((size_t)b * NH + h) * DKH * SEQ;

  v8f oc[4] = {zero8(), zero8(), zero8(), zero8()};
  float mrow[8], lrow[8];
#pragma unroll
  for (int r = 0; r < 8; ++r) { mrow[r] = -3.0e30f; lrow[r] = 0.f; }

  for (int kb = 0; kb < 26; ++kb) {             // ceil(816/32)
    __syncthreads();
    for (int idx = (int)threadIdx.x; idx < 256; idx += 96) {   // K tile 32x64
      int row = idx >> 3, ch = idx & 7;
      int key = kb * 32 + row; if (key > SEQ - 1) key = SEQ - 1;
      async_b128(sK + row * 64 + ch * 8, kbase + (size_t)key * DIM + ch * 8);
    }
    for (int idx = (int)threadIdx.x; idx < 256; idx += 96) {   // Vt tile 64x32
      int row = idx >> 2, ch = idx & 3;
      int ks = kb * 32 + ch * 8; if (ks > SEQ - 8) ks = SEQ - 8;
      async_b128(sV + row * 32 + ch * 8, vbase + (size_t)row * SEQ + ks);
    }
    wait_async0();
    __syncthreads();

    v8f s0 = zero8(), s1 = zero8();
    s0 = WMMA_BF16(aq0, frag_b(sK, 64), s0);
    s0 = WMMA_BF16(aq1, frag_b(sK + 32, 64), s0);
    s1 = WMMA_BF16(aq0, frag_b(sK + 16 * 64, 64), s1);
    s1 = WMMA_BF16(aq1, frag_b(sK + 16 * 64 + 32, 64), s1);

    int kc0 = kb * 32 + (int)n, kc1 = kc0 + 16;
    __bf16* pw = &sP[wave][0];
#pragma unroll
    for (int r = 0; r < 8; ++r) {
      float a0 = s0[r] * 0.125f; if (kc0 >= SEQ) a0 = -3.0e30f;
      float a1 = s1[r] * 0.125f; if (kc1 >= SEQ) a1 = -3.0e30f;
      float mx = redmax16(fmaxf(a0, a1));
      float newm = fmaxf(mrow[r], mx);
      float scl = __expf(mrow[r] - newm);
      mrow[r] = newm;
      float p0 = __expf(a0 - newm);
      float p1 = __expf(a1 - newm);
      lrow[r] = lrow[r] * scl + redsum16(p0 + p1);
      oc[0][r] *= scl; oc[1][r] *= scl; oc[2][r] *= scl; oc[3][r] *= scl;
      int prow = (r + 8 * (int)hs) * 32;
      pw[prow + (int)n] = (__bf16)p0;
      pw[prow + 16 + (int)n] = (__bf16)p1;
    }
    asm volatile("s_wait_dscnt 0x0" ::: "memory");  // same-wave LDS RAW fence
    v16bf ap = frag_a(&sP[wave][0], 32);
#pragma unroll
    for (int ns = 0; ns < 4; ++ns) {
      v16bf bv = frag_b(sV + ns * 16 * 32, 32);
      oc[ns] = WMMA_BF16(ap, bv, oc[ns]);
    }
  }

#pragma unroll
  for (int r = 0; r < 8; ++r) {
    float inv = 1.0f / lrow[r];
    int qrow = qt * 16 + r + 8 * (int)hs;
#pragma unroll
    for (int ns = 0; ns < 4; ++ns)
      Ob[((size_t)b * SEQ + qrow) * DIM + h * DKH + ns * 16 + (int)n] =
          (__bf16)(oc[ns][r] * inv);
  }
}

// ---------------- Kernel 6: output projection (fp32 out + bias) ----------------
// Block: 256 threads (8 waves). Tile 128 rows x 64 cols. M = 32*816 = 26112.

__global__ __launch_bounds__(256) void outproj(
    const __bf16* __restrict__ Ob, const __bf16* __restrict__ Wo,
    const float* __restrict__ bias, float* __restrict__ out) {
  __shared__ __attribute__((aligned(16))) __bf16 sA[128 * 32];
  __shared__ __attribute__((aligned(16))) __bf16 sB[64 * 32];
  int mt = blockIdx.x;   // 0..203
  int nt = blockIdx.y;   // 0..7
  int wave = (int)(threadIdx.x >> 5);
  v8f acc[4] = {zero8(), zero8(), zero8(), zero8()};

  for (int kc = 0; kc < 16; ++kc) {
    int kk = kc * 32;
    __syncthreads();
    for (int idx = (int)threadIdx.x; idx < 512; idx += 256) {
      int row = idx >> 2, ch = idx & 3;
      async_b128(sA + row * 32 + ch * 8,
                 Ob + (size_t)(mt * 128 + row) * DIM + kk + ch * 8);
    }
    {
      int idx = (int)threadIdx.x;
      int row = idx >> 2, ch = idx & 3;
      async_b128(sB + row * 32 + ch * 8,
                 Wo + (size_t)(nt * 64 + row) * DIM + kk + ch * 8);
    }
    wait_async0();
    __syncthreads();
    v16bf a = frag_a(sA + wave * 16 * 32, 32);
#pragma unroll
    for (int ns = 0; ns < 4; ++ns) {
      v16bf bm = frag_b(sB + ns * 16 * 32, 32);
      acc[ns] = WMMA_BF16(a, bm, acc[ns]);
    }
  }
  unsigned lane = threadIdx.x & 31u, n = lane & 15u, hs = lane >> 4u;
#pragma unroll
  for (int ns = 0; ns < 4; ++ns)
#pragma unroll
    for (int r = 0; r < 8; ++r) {
      int row = mt * 128 + wave * 16 + r + 8 * (int)hs;
      int col = nt * 64 + ns * 16 + (int)n;
      out[(size_t)row * DIM + col] = acc[ns][r] + bias[col];
    }
}

// ---------------- Host launcher ----------------

extern "C" void kernel_launch(void* const* d_in, const int* in_sizes, int n_in,
                              void* d_out, int out_size, void* d_ws, size_t ws_size,
                              hipStream_t stream) {
  (void)in_sizes; (void)n_in; (void)out_size; (void)ws_size;
  const float* X   = (const float*)d_in[0];
  const float* wk  = (const float*)d_in[1];
  const float* wq  = (const float*)d_in[2];
  const float* wv  = (const float*)d_in[3];
  const float* qkw = (const float*)d_in[4];
  const float* qkb = (const float*)d_in[5];
  const float* qqw = (const float*)d_in[6];
  const float* qqb = (const float*)d_in[7];
  const float* qvw = (const float*)d_in[8];
  const float* qvb = (const float*)d_in[9];
  const float* ow  = (const float*)d_in[10];
  const float* obv = (const float*)d_in[11];
  float* out = (float*)d_out;

  char* p = (char*)d_ws;
  auto alloc = [&](size_t bytes) -> char* {
    char* r = p;
    p += (bytes + 255) & ~(size_t)255;
    return r;
  };
  __bf16* Xb = (__bf16*)alloc((size_t)BSZ * SEQ * DIM * 2);       // bf16 input
  __bf16* Wc = (__bf16*)alloc((size_t)3 * 9 * DIM * DIM * 2);     // conv w per tap
  __bf16* Wl = (__bf16*)alloc((size_t)3 * DIM * DIM * 2);         // linear w
  __bf16* Wo = (__bf16*)alloc((size_t)DIM * DIM * 2);             // out proj w
  __bf16* Kb = (__bf16*)alloc((size_t)BSZ * SEQ * DIM * 2);       // K  [b][s][d]
  __bf16* Qb = (__bf16*)alloc((size_t)BSZ * SEQ * DIM * 2);       // Q  [b][s][d]
  __bf16* Vt = (__bf16*)alloc((size_t)BSZ * NH * DKH * SEQ * 2);  // V^T [b][h][d][s]
  __bf16* Obuf = (__bf16*)alloc((size_t)BSZ * SEQ * DIM * 2);     // attn out

  cast_input<<<4096, 256, 0, stream>>>(X, Xb);
  pack_weights<<<2048, 256, 0, stream>>>(wk, wq, wv, qkw, qqw, qvw, ow, Wc, Wl, Wo);
  qlin<<<dim3(64, 32, 3), 32, 0, stream>>>(Xb, Wl, qkb, qqb, qvb, Kb, Qb, Vt);
  convqkv<<<dim3(7, 8, BSZ * 3), 224, 0, stream>>>(Xb, Wc, Kb, Qb, Vt);
  attn<<<dim3(17, BSZ * NH), 96, 0, stream>>>(Qb, Kb, Vt, Obuf);
  outproj<<<dim3(204, 8), 256, 0, stream>>>(Obuf, Wo, obv, out);
}
